// NearestUpsampleBlock_62062277427558
// MI455X (gfx1250) — compile-verified, matrix operand order checked by reading, not run on hardware
//
#include <hip/hip_runtime.h>

typedef __attribute__((ext_vector_type(16))) __bf16 v16bf;
typedef __attribute__((ext_vector_type(8)))  __bf16 v8bf;
typedef __attribute__((ext_vector_type(4)))  __bf16 v4bf;
typedef __attribute__((ext_vector_type(8)))  float  v8f;
typedef __attribute__((ext_vector_type(4)))  float  v4f;

static constexpr int Bc   = 4;
static constexpr int Nc   = 8192;
static constexpr int Sc   = 2048;
static constexpr int D1c  = 256;
static constexpr int D2c  = 512;
static constexpr int COUTc = 256;
static constexpr int Kc   = 8;            // nsample
static constexpr int KDIM = D1c + D2c;    // 768 (GEMM K)
static constexpr int Mc   = Bc * Nc;      // 32768 (GEMM M)

#define PPB 32   // points per block
#define TPP 8    // threads per point

// ---------------------------------------------------------------------------
// Kernel 1: kNN (top-8 of 2048) + inverse-distance interpolation, packs the
// bf16 activation matrix A[M][768] = [points1_bf16 | interpolated_bf16].
// ---------------------------------------------------------------------------
__global__ __launch_bounds__(256) void knn_interp_pack(
    const float* __restrict__ xyz1, const float* __restrict__ xyz2,
    const float* __restrict__ points1, const float* __restrict__ points2,
    __bf16* __restrict__ A)
{
  __shared__ float s_xyz[Sc][4];          // x,y,z,|.|^2  (32 KB)
  __shared__ float s_md[PPB][TPP * Kc];   // merge dists  (8 KB)
  __shared__ int   s_mi[PPB][TPP * Kc];   // merge idx    (8 KB)
  __shared__ float s_w[PPB][Kc];
  __shared__ int   s_i[PPB][Kc];

  const int tid   = threadIdx.x;
  const int pbase = blockIdx.x * PPB;     // global row base in [0, M)
  const int b     = pbase / Nc;           // batch (blocks never straddle)

  // Phase A: stage xyz2 for this batch (+ squared norms) into LDS
  for (int s = tid; s < Sc; s += 256) {
    float x = xyz2[(b * Sc + s) * 3 + 0];
    float y = xyz2[(b * Sc + s) * 3 + 1];
    float z = xyz2[(b * Sc + s) * 3 + 2];
    s_xyz[s][0] = x; s_xyz[s][1] = y; s_xyz[s][2] = z;
    s_xyz[s][3] = x * x + y * y + z * z;
  }
  __syncthreads();

  const int g   = tid >> 3;   // point group 0..31
  const int j   = tid & 7;    // lane within group
  const int row = pbase + g;  // global row index

  // Phase B: each of 8 lanes scans S/8 candidates, register-sorted top-8
  {
    const float x1 = xyz1[row * 3 + 0];
    const float y1 = xyz1[row * 3 + 1];
    const float z1 = xyz1[row * 3 + 2];
    const float sq1 = x1 * x1 + y1 * y1 + z1 * z1;

    float bd[Kc]; int bi[Kc];
#pragma unroll
    for (int t = 0; t < Kc; ++t) { bd[t] = 3.0e38f; bi[t] = 0; }

    for (int s = j; s < Sc; s += TPP) {
      float d = sq1 + s_xyz[s][3]
              - 2.0f * (x1 * s_xyz[s][0] + y1 * s_xyz[s][1] + z1 * s_xyz[s][2]);
      if (d < bd[Kc - 1]) {
        bd[Kc - 1] = d; bi[Kc - 1] = s;
#pragma unroll
        for (int t = Kc - 1; t > 0; --t) {   // one bubble pass restores order
          if (bd[t] < bd[t - 1]) {
            float td = bd[t]; bd[t] = bd[t - 1]; bd[t - 1] = td;
            int   ti = bi[t]; bi[t] = bi[t - 1]; bi[t - 1] = ti;
          }
        }
      }
    }
#pragma unroll
    for (int t = 0; t < Kc; ++t) { s_md[g][j * Kc + t] = bd[t]; s_mi[g][j * Kc + t] = bi[t]; }
  }
  __syncthreads();

  // Phase B2: group leader merges 64 -> 8, builds normalized 1/d weights
  if (j == 0) {
    float wsum = 0.0f;
#pragma unroll 1
    for (int t = 0; t < Kc; ++t) {
      float best = 3.0e38f; int bestq = 0;
      for (int q = 0; q < TPP * Kc; ++q) {
        float d = s_md[g][q];
        if (d < best) { best = d; bestq = q; }
      }
      s_md[g][bestq] = 3.0e38f;
      float w = 1.0f / (best + 1e-8f);
      s_w[g][t] = w; s_i[g][t] = s_mi[g][bestq];
      wsum += w;
    }
    const float inv = 1.0f / wsum;
#pragma unroll
    for (int t = 0; t < Kc; ++t) s_w[g][t] *= inv;
  }
  __syncthreads();

  // Phase C: weighted gather of neighbor features -> bf16 A[:, 256:768]
  {
    float w[Kc]; long ofs[Kc];
#pragma unroll
    for (int t = 0; t < Kc; ++t) {
      w[t]   = s_w[g][t];
      ofs[t] = (long)(b * Sc + s_i[g][t]) * (D2c / 4);  // float4 units
    }
    const v4f* p2 = (const v4f*)points2;
#pragma unroll 2
    for (int q = j; q < D2c / 4; q += TPP) {
      v4f acc = {0.f, 0.f, 0.f, 0.f};
#pragma unroll
      for (int t = 0; t < Kc; ++t) {
        v4f v = p2[ofs[t] + q];
        acc += v * w[t];
      }
      v4bf o;
      o[0] = (__bf16)acc[0]; o[1] = (__bf16)acc[1];
      o[2] = (__bf16)acc[2]; o[3] = (__bf16)acc[3];
      *(v4bf*)(A + (long)row * KDIM + D1c + q * 4) = o;
    }
  }

  // Phase D: pack the block's 32 rows of points1 -> bf16 A[:, 0:256]
  {
    const v4f* p1 = (const v4f*)points1;
    for (int v = tid; v < PPB * (D1c / 4); v += 256) {
      int r = v >> 6;            // D1/4 = 64 float4 per row
      int q = v & 63;
      v4f x = p1[(long)(pbase + r) * (D1c / 4) + q];
      v4bf o;
      o[0] = (__bf16)x[0]; o[1] = (__bf16)x[1];
      o[2] = (__bf16)x[2]; o[3] = (__bf16)x[3];
      *(v4bf*)(A + (long)(pbase + r) * KDIM + q * 4) = o;
    }
  }
}

// ---------------------------------------------------------------------------
// Kernel 2: conv weight fp32 -> bf16 (row-major [COUT][768], same layout as A)
// ---------------------------------------------------------------------------
__global__ __launch_bounds__(256) void pack_w(const float* __restrict__ Wf,
                                              __bf16* __restrict__ Wb)
{
  int i = blockIdx.x * 256 + threadIdx.x;   // float4 index
  if (i < COUTc * KDIM / 4) {
    v4f x = ((const v4f*)Wf)[i];
    v4bf o;
    o[0] = (__bf16)x[0]; o[1] = (__bf16)x[1];
    o[2] = (__bf16)x[2]; o[3] = (__bf16)x[3];
    ((v4bf*)Wb)[i] = o;
  }
}

// ---------------------------------------------------------------------------
// Kernel 3: out[M][256] = A[M][768] x Wb^T + bias, bf16 WMMA, fp32 accumulate.
// 8 waves / block; wave owns a 16x64 tile (4 accumulators); 24 K-chunks.
// Fragment layout per ISA 7.12.2 (16-bit A/B): lane-half selects
// K in {8h..8h+7} U {8h+16..8h+23} within each 32-wide K chunk.
// ---------------------------------------------------------------------------
__global__ __launch_bounds__(256) void wmma_gemm(
    const __bf16* __restrict__ A, const __bf16* __restrict__ Wb,
    const float* __restrict__ bias, float* __restrict__ out)
{
  const int lane  = threadIdx.x & 31;
  const int wave  = threadIdx.x >> 5;
  const int mtile = blockIdx.x * 8 + wave;   // 16-row tile (2048 total)
  const int n0    = blockIdx.y * 64;         // 64-column group
  const int half  = lane >> 4;
  const int l15   = lane & 15;

  const long arow = (long)(mtile * 16 + l15) * KDIM;

  // accumulators pre-loaded with bias (D-layout: N = l15, same for all VGPRs)
  v8f acc[4];
#pragma unroll
  for (int t = 0; t < 4; ++t) {
    float bv = bias[n0 + t * 16 + l15];
#pragma unroll
    for (int v = 0; v < 8; ++v) acc[t][v] = bv;
  }

#pragma unroll 1
  for (int kc = 0; kc < KDIM / 32; ++kc) {
    const int kb = kc * 32 + half * 8;
    v8bf alo = *(const v8bf*)(A + arow + kb);
    v8bf ahi = *(const v8bf*)(A + arow + kb + 16);
    v16bf afrag = __builtin_shufflevector(alo, ahi,
        0, 1, 2, 3, 4, 5, 6, 7, 8, 9, 10, 11, 12, 13, 14, 15);
#pragma unroll
    for (int t = 0; t < 4; ++t) {
      const long wrow = (long)(n0 + t * 16 + l15) * KDIM;
      v8bf blo = *(const v8bf*)(Wb + wrow + kb);
      v8bf bhi = *(const v8bf*)(Wb + wrow + kb + 16);
      v16bf bfrag = __builtin_shufflevector(blo, bhi,
          0, 1, 2, 3, 4, 5, 6, 7, 8, 9, 10, 11, 12, 13, 14, 15);
      acc[t] = __builtin_amdgcn_wmma_f32_16x16x32_bf16(
          false, afrag, false, bfrag, (short)0, acc[t], false, false);
    }
  }

  // store (D layout: lanes 0-15 -> M=v, lanes 16-31 -> M=v+8)
#pragma unroll
  for (int t = 0; t < 4; ++t) {
    const int o = n0 + t * 16 + l15;
#pragma unroll
    for (int v = 0; v < 8; ++v) {
      const int m = mtile * 16 + v + 8 * half;
      out[(long)m * COUTc + o] = acc[t][v];
    }
  }
}

// ---------------------------------------------------------------------------
extern "C" void kernel_launch(void* const* d_in, const int* in_sizes, int n_in,
                              void* d_out, int out_size, void* d_ws, size_t ws_size,
                              hipStream_t stream) {
  const float* xyz1    = (const float*)d_in[0];
  const float* xyz2    = (const float*)d_in[1];
  const float* points1 = (const float*)d_in[2];
  const float* points2 = (const float*)d_in[3];
  const float* conv_w  = (const float*)d_in[4];
  const float* conv_b  = (const float*)d_in[5];
  // d_in[6] = nsample (==8), baked into Kc

  __bf16* A  = (__bf16*)d_ws;                                       // 48 MiB
  __bf16* Wb = (__bf16*)((char*)d_ws + (size_t)Mc * KDIM * 2);      // +384 KiB
  float* out = (float*)d_out;

  knn_interp_pack<<<Mc / PPB, 256, 0, stream>>>(xyz1, xyz2, points1, points2, A);
  pack_w<<<(COUTc * KDIM / 4 + 255) / 256, 256, 0, stream>>>(conv_w, Wb);
  wmma_gemm<<<dim3(Mc / 16 / 8, COUTc / 64), 256, 0, stream>>>(A, Wb, conv_b, out);
}
